// QuantLinear4bit_77249281786460
// MI455X (gfx1250) — compile-verified, matrix operand order checked by reading, not run on hardware
//
#include <hip/hip_runtime.h>

typedef __attribute__((ext_vector_type(16))) _Float16 v16h;
typedef __attribute__((ext_vector_type(2)))  _Float16 v2h;
typedef __attribute__((ext_vector_type(8)))  float    v8f;

#define IN_F   4096
#define OUT_F  11008
#define KBLK   32          // K_BLOCKS
#define GS     128         // group size

#define BM 128
#define BN 128
#define BK 64
#define LDA 72             // LDS row stride in halves (64 + 8 pad, keeps 16B align)
#define LDB 72

union Frag {
    v16h  h;
    uint4 u[2];
};

static __device__ __forceinline__ unsigned hbits(v2h h) {
    return __builtin_bit_cast(unsigned, h);
}

// Dequant two nibbles of one packed byte -> two f16 weights:
//   as_f16(0x6400 | q) == 1024 + q  (exact),  w = ((1024+q) - (1024+z)) * s
static __device__ __forceinline__ v2h dq2(int q, v2h z2, v2h s2) {
    unsigned p = ((unsigned)q & 15u) | (((unsigned)q << 12) & 0xF0000u) | 0x64006400u;
    v2h qh = __builtin_bit_cast(v2h, p);
    return (qh - z2) * s2;           // v_pk_add_f16(neg) + v_pk_mul_f16
}

__global__ __launch_bounds__(256)
void qlinear4_wmma_f16(const float* __restrict__ x,        // [M, 4096]
                       const int*   __restrict__ qweight,  // [11008, 32, 64]
                       const float* __restrict__ scales,   // [11008, 32]
                       const int*   __restrict__ qzeros,   // [11008, 16]
                       float*       __restrict__ out,      // [M, 11008]
                       int M)
{
    __shared__ _Float16 lds_a[2][BM * LDA];   // double-buffered f16 tiles
    __shared__ _Float16 lds_b[2][BN * LDB];

    const int t    = threadIdx.x;
    const int lane = t & 31;
    const int wave = t >> 5;
    const int wm   = wave & 3;          // 4 waves along M -> 32 rows each
    const int wn   = wave >> 2;         // 2 waves along N -> 64 cols each
    const int lrow = lane & 15;
    const int khalf = (lane >> 4) << 3; // 0 or 8

    const int n0 = blockIdx.x * BN;
    const int m0 = blockIdx.y * BM;

    // A loader: each thread covers 8 contiguous floats -> one b128 LDS store; 4 passes
    const int a_c8 = t & 7;
    const int a_r0 = t >> 3;
    const float* xp0 = x + (size_t)(m0 + a_r0) * IN_F + a_c8 * 8;

    // B loader: 2 threads per n-row; each covers 16 packed bytes (32 k) per stage.
    // qweight is [n][kb][64] ints => a stage's span starts at n*2048 + k0/2 (linear in k0).
    const int b_n = t >> 1;
    const int b_g = t & 1;
    const int bn_g = n0 + b_n;
    const int* qp0 = qweight + (size_t)bn_g * (KBLK * GS / 2) + b_g * 16;
    const float* sp = scales + (size_t)bn_g * KBLK;
    const int*   zp = qzeros + (size_t)bn_g * (KBLK / 2);

    v8f acc[2][4] = {};

    auto stage_load = [&](int buf, int k0) {
        // ---- A: 128x64 f32 -> f16 ----
        #pragma unroll
        for (int r = 0; r < 4; ++r) {
            const float* xp = xp0 + (size_t)r * 32 * IN_F + k0;
            float4 v0 = *(const float4*)(xp);
            float4 v1 = *(const float4*)(xp + 4);
            uint4 st;
            st.x = __builtin_bit_cast(unsigned, __builtin_amdgcn_cvt_pkrtz(v0.x, v0.y));
            st.y = __builtin_bit_cast(unsigned, __builtin_amdgcn_cvt_pkrtz(v0.z, v0.w));
            st.z = __builtin_bit_cast(unsigned, __builtin_amdgcn_cvt_pkrtz(v1.x, v1.y));
            st.w = __builtin_bit_cast(unsigned, __builtin_amdgcn_cvt_pkrtz(v1.z, v1.w));
            *(uint4*)&lds_a[buf][(a_r0 + r * 32) * LDA + a_c8 * 8] = st;
        }
        // ---- B: dequant 128n x 64k -> f16 ----
        {
            const int   kb = k0 >> 7;
            const float s  = sp[kb];
            const int   qz = zp[kb >> 1];
            const int   z  = (kb & 1) ? ((qz >> 4) & 15) : (qz & 15);
            const _Float16 sh = (_Float16)s;
            const _Float16 zh = (_Float16)(float)(1024 + z);  // exact in f16
            const v2h s2 = { sh, sh };
            const v2h z2 = { zh, zh };
            const int* qp = qp0 + (k0 >> 1);
            #pragma unroll
            for (int i = 0; i < 4; ++i) {
                int4 q = *(const int4*)(qp + i * 4);
                uint4 st;
                st.x = hbits(dq2(q.x, z2, s2));
                st.y = hbits(dq2(q.y, z2, s2));
                st.z = hbits(dq2(q.z, z2, s2));
                st.w = hbits(dq2(q.w, z2, s2));
                *(uint4*)&lds_b[buf][b_n * LDB + (b_g * 16 + i * 4) * 2] = st;
            }
        }
    };

    auto compute = [&](int buf) {
        #pragma unroll
        for (int ks = 0; ks < BK; ks += 32) {
            Frag a[2], b[4];
            #pragma unroll
            for (int mi = 0; mi < 2; ++mi) {
                const _Float16* p = &lds_a[buf][(wm * 32 + mi * 16 + lrow) * LDA + ks + khalf];
                a[mi].u[0] = *(const uint4*)(p);        // K = kbase .. +7
                a[mi].u[1] = *(const uint4*)(p + 16);   // K = kbase+16 .. +23
            }
            #pragma unroll
            for (int ni = 0; ni < 4; ++ni) {
                const _Float16* p = &lds_b[buf][(wn * 64 + ni * 16 + lrow) * LDB + ks + khalf];
                b[ni].u[0] = *(const uint4*)(p);
                b[ni].u[1] = *(const uint4*)(p + 16);
            }
            #pragma unroll
            for (int mi = 0; mi < 2; ++mi)
                #pragma unroll
                for (int ni = 0; ni < 4; ++ni)
                    acc[mi][ni] = __builtin_amdgcn_wmma_f32_16x16x32_f16(
                        false, a[mi].h, false, b[ni].h,
                        (short)0, acc[mi][ni], false, false);
        }
    };

    stage_load(0, 0);

    int buf = 0;
    // branch-free steady state: always load next stage, compute current
    for (int k0 = 0; k0 < IN_F - BK; k0 += BK) {
        __syncthreads();
        __builtin_prefetch(xp0 + (k0 + 2 * BK), 0, 0);
        stage_load(buf ^ 1, k0 + BK);
        compute(buf);
        buf ^= 1;
    }
    __syncthreads();
    compute(buf);        // peeled final stage

    // ---- writeback: VGPR i -> M = base + 8*(lane>=16) + i, N = lane&15 ----
    const int orow = (lane >> 4) << 3;
    #pragma unroll
    for (int mi = 0; mi < 2; ++mi) {
        #pragma unroll
        for (int ni = 0; ni < 4; ++ni) {
            const int m = m0 + wm * 32 + mi * 16 + orow;
            const int n = n0 + wn * 64 + ni * 16 + lrow;
            #pragma unroll
            for (int i = 0; i < 8; ++i) {
                out[(size_t)(m + i) * OUT_F + n] = acc[mi][ni][i];
            }
        }
    }
}

extern "C" void kernel_launch(void* const* d_in, const int* in_sizes, int n_in,
                              void* d_out, int out_size, void* d_ws, size_t ws_size,
                              hipStream_t stream) {
    (void)n_in; (void)out_size; (void)d_ws; (void)ws_size;
    const float* x       = (const float*)d_in[0];
    const int*   qweight = (const int*)  d_in[1];
    const float* scales  = (const float*)d_in[2];
    const int*   qzeros  = (const int*)  d_in[3];
    float*       out     = (float*)      d_out;

    const int M = in_sizes[0] / IN_F;   // 2048
    dim3 grid(OUT_F / BN, M / BM);      // (86, 16)
    qlinear4_wmma_f16<<<grid, 256, 0, stream>>>(x, qweight, scales, qzeros, out, M);
}
